// EntityGuidedCrossAttention_76081050681504
// MI455X (gfx1250) — matrix operand order repaired and checked by reading, not compiled, not run on hardware
//
#include <hip/hip_runtime.h>
#include <hip/hip_bf16.h>

// ---------------------------------------------------------------------------
// Problem constants (from the reference)
// ---------------------------------------------------------------------------
#define N_CLASSES 512
#define K_SHOTS   64
#define DFEAT     1024
#define NK        (N_CLASSES * K_SHOTS)   // 32768

typedef unsigned short u16;
typedef unsigned int   u32;

typedef __attribute__((ext_vector_type(16))) __bf16 v16bf;
typedef __attribute__((ext_vector_type(8)))  float  v8f;

union Frag16 {           // 8 dwords == 16 bf16 == one WMMA A/B operand
    u32   u[8];
    v16bf v;
};

// float -> bf16 (round to nearest even, ignoring NaN edge cases)
__device__ __forceinline__ u16 f2bf(float x) {
    u32 u = __float_as_uint(x);
    u32 r = (u + 0x7FFFu + ((u >> 16) & 1u)) >> 16;
    return (u16)r;
}
__device__ __forceinline__ float bf2f(u16 b) {
    return __uint_as_float(((u32)b) << 16);
}

// ---------------------------------------------------------------------------
// 1) elementwise fp32 -> bf16 (x4 vectorized; element counts are mult. of 1024)
// ---------------------------------------------------------------------------
__global__ __launch_bounds__(256) void to_bf16_x4(const float4* __restrict__ in,
                                                  u32* __restrict__ out /* 2 bf16 per u32 */,
                                                  int n4) {
    int i = blockIdx.x * 256 + threadIdx.x;
    if (i >= n4) return;
    float4 v = in[i];
    u32 lo = (u32)f2bf(v.x) | ((u32)f2bf(v.y) << 16);
    u32 hi = (u32)f2bf(v.z) | ((u32)f2bf(v.w) << 16);
    out[2 * i]     = lo;
    out[2 * i + 1] = hi;
}

// ---------------------------------------------------------------------------
// 2) W [D,D] fp32 row-major  ->  WT [D,D] bf16 row-major  (WT[d][n] = W[n][d])
// ---------------------------------------------------------------------------
__global__ __launch_bounds__(256) void transpose_to_bf16(const float* __restrict__ W,
                                                         u16* __restrict__ WT,
                                                         int Dm) {
    __shared__ float tile[32][33];
    int bx = blockIdx.x * 32;   // d-tile
    int by = blockIdx.y * 32;   // n-tile
    int tx = threadIdx.x & 31;
    int ty = threadIdx.x >> 5;  // 0..7
    for (int i = 0; i < 32; i += 8)
        tile[ty + i][tx] = W[(size_t)(by + ty + i) * Dm + (bx + tx)];
    __syncthreads();
    for (int i = 0; i < 32; i += 8)
        WT[(size_t)(bx + ty + i) * Dm + (by + tx)] = f2bf(tile[tx][ty + i]);
}

// ---------------------------------------------------------------------------
// 3) bf16 WMMA GEMM:  C[M,N] = A[M,K] @ B[K,N] + bias,  fp32 accumulate.
//    A, B bf16 row-major.  C fp32 or bf16 (out_bf16 flag).
//    Block tile 128x128, BK=32, 8 waves (2x4), wave tile 64x32 (4x2 wmmas).
//    Software-pipelined: next K-tile global loads are in flight during WMMAs.
//    M, N, K must be multiples of 128/128/32 (true for all uses here).
// ---------------------------------------------------------------------------
#define BM 128
#define BN 128
#define BK 32
#define APITCH 40   // bf16 elements per A row in LDS (80 B: 16B aligned, conflict-free)
#define BPITCH 17   // dwords per B column in LDS

__global__ __launch_bounds__(256) void gemm_bf16_wmma(const u16* __restrict__ A,
                                                      const u16* __restrict__ B,
                                                      const float* __restrict__ bias,
                                                      void* __restrict__ C,
                                                      int M, int N, int K,
                                                      int out_bf16) {
    __shared__ u16 lA[BM][APITCH];       // A tile, row-major bf16
    __shared__ u32 lB[BN][BPITCH];       // B tile: [col][k-pair] interleaved dwords

    const int tid  = threadIdx.x;
    const int lane = tid & 31;
    const int wave = tid >> 5;           // 0..7
    const int wm0  = (wave >> 2) * 64;   // wave M origin inside block tile
    const int wn0  = (wave & 3) * 32;    // wave N origin inside block tile

    const int bm0 = blockIdx.x * BM;
    const int bn0 = blockIdx.y * BN;

    v8f acc[4][2];
    for (int m = 0; m < 4; ++m)
        for (int n = 0; n < 2; ++n)
            for (int e = 0; e < 8; ++e)
                acc[m][n][e] = 0.0f;

    // ---- staging thread mapping ----
    // A: 512 x 16B chunks; thread handles chunks tid and tid+256
    //    (row1 = row0 + 64, same column group)
    const int arow = tid >> 2;           // 0..63
    const int acol = (tid & 3) * 8;      // 0,8,16,24 (bf16 elems)
    // B: thread handles k-pair row bp (k = 2bp, 2bp+1), 8 columns at bc0
    const int bp  = tid & 15;
    const int bc0 = (tid >> 4) * 8;

    const u16* aP0 = A + (size_t)(bm0 + arow) * K + acol;
    const u16* aP1 = aP0 + (size_t)64 * K;
    const u16* bP  = B + (size_t)(2 * bp) * N + (bn0 + bc0);

    // ---- prologue: load K-tile 0 into registers ----
    uint4 ra0 = *(const uint4*)aP0;
    uint4 ra1 = *(const uint4*)aP1;
    uint4 rb0 = *(const uint4*)bP;
    uint4 rb1 = *(const uint4*)(bP + N);
    aP0 += BK; aP1 += BK; bP += (size_t)BK * N;

    auto stage = [&]() {
        *(uint4*)&lA[arow][acol]      = ra0;
        *(uint4*)&lA[arow + 64][acol] = ra1;
        u32 e0[4] = {rb0.x, rb0.y, rb0.z, rb0.w};
        u32 e1[4] = {rb1.x, rb1.y, rb1.z, rb1.w};
        for (int i = 0; i < 4; ++i) {
            lB[bc0 + 2 * i][bp]     = (e0[i] & 0xFFFFu) | (e1[i] << 16);
            lB[bc0 + 2 * i + 1][bp] = (e0[i] >> 16) | (e1[i] & 0xFFFF0000u);
        }
    };

    auto compute = [&]() {
        // A 16x32 bf16 operand: lanes 0-15 row m=lane, K 0-7 (v0-3) / 16-23 (v4-7)
        //                       lanes 16-31 row m=lane-16, K 8-15 / 24-31
        Frag16 fragA[4];
        const int klo = (lane < 16) ? 0 : 8;
        for (int sm = 0; sm < 4; ++sm) {
            int mrow = wm0 + sm * 16 + (lane & 15);
            for (int j = 0; j < 4; ++j) {
                fragA[sm].u[j]     = *(const u32*)&lA[mrow][klo + 2 * j];
                fragA[sm].u[j + 4] = *(const u32*)&lA[mrow][16 + klo + 2 * j];
            }
        }
        // B 32x16 bf16 operand: VGPR j, lane L -> col = L%16, k-pair = j + (j&4) + (L<16?0:4)
        Frag16 fragB[2];
        const int rbase = (lane < 16) ? 0 : 4;
        for (int sn = 0; sn < 2; ++sn) {
            int col = wn0 + sn * 16 + (lane & 15);
            for (int j = 0; j < 8; ++j)
                fragB[sn].u[j] = lB[col][rbase + j + (j & 4)];
        }
        for (int sm = 0; sm < 4; ++sm)
            for (int sn = 0; sn < 2; ++sn)
                acc[sm][sn] = __builtin_amdgcn_wmma_f32_16x16x32_bf16(
                    false, fragA[sm].v, false, fragB[sn].v,
                    (short)0, acc[sm][sn], false, false);
    };

    // ---- steady state: no conditionals; next-tile loads overlap the WMMAs ----
    for (int k0 = 0; k0 < K - BK; k0 += BK) {
        stage();
        __syncthreads();
        uint4 na0 = *(const uint4*)aP0;          // in flight during compute()
        uint4 na1 = *(const uint4*)aP1;
        uint4 nb0 = *(const uint4*)bP;
        uint4 nb1 = *(const uint4*)(bP + N);
        aP0 += BK; aP1 += BK; bP += (size_t)BK * N;
        __builtin_prefetch(aP0, 0, 1);           // speculative; OOB is dropped
        __builtin_prefetch(bP, 0, 1);
        compute();
        __syncthreads();
        ra0 = na0; ra1 = na1; rb0 = nb0; rb1 = nb1;
    }
    // ---- epilogue tile ----
    stage();
    __syncthreads();
    compute();

    // ---- C/D layout: VGPR v, lane L -> row = v + (L<16?0:8), col = L%16 ----
    const int rhalf = (lane < 16) ? 0 : 8;
    for (int sm = 0; sm < 4; ++sm) {
        for (int sn = 0; sn < 2; ++sn) {
            int ccol = bn0 + wn0 + sn * 16 + (lane & 15);
            float bv = bias ? bias[ccol] : 0.0f;
            int rbase = bm0 + wm0 + sm * 16 + rhalf;
            for (int v = 0; v < 8; ++v) {
                float val = acc[sm][sn][v] + bv;
                size_t idx = (size_t)(rbase + v) * N + ccol;
                if (out_bf16) ((u16*)C)[idx]   = f2bf(val);
                else          ((float*)C)[idx] = val;
            }
        }
    }
}

// ---------------------------------------------------------------------------
// 4) per-class attention: 512 blocks, 256 threads.
//    scores_j = <Q[c], Kp[c*64+j]> / 32 ; softmax over 64 ; attended = w @ Vp
// ---------------------------------------------------------------------------
__global__ __launch_bounds__(256) void class_attention(const float* __restrict__ Q,
                                                       const u16* __restrict__ Kp,
                                                       const u16* __restrict__ Vp,
                                                       u16* __restrict__ att) {
    const int c   = blockIdx.x;
    const int tid = threadIdx.x;
    __shared__ float sQ[DFEAT];
    __shared__ float sS[K_SHOTS];
    __shared__ float red[256];

    for (int i = tid; i < DFEAT; i += 256)
        sQ[i] = Q[(size_t)c * DFEAT + i];
    __syncthreads();

    // dot products: 4 threads per support sample, 256 elems each
    {
        int j = tid >> 2, q = tid & 3;
        const u16* kr = Kp + (size_t)(c * K_SHOTS + j) * DFEAT + q * 256;
        const float* qr = sQ + q * 256;
        float s = 0.0f;
        for (int i = 0; i < 256; ++i) s += qr[i] * bf2f(kr[i]);
        red[tid] = s;
    }
    __syncthreads();
    if ((tid & 3) == 0)
        sS[tid >> 2] = (red[tid] + red[tid + 1] + red[tid + 2] + red[tid + 3]) * 0.03125f;
    __syncthreads();

    if (tid == 0) {  // 64-way softmax, trivial cost
        float m = sS[0];
        for (int j = 1; j < K_SHOTS; ++j) m = fmaxf(m, sS[j]);
        float sum = 0.0f;
        for (int j = 0; j < K_SHOTS; ++j) { float e = __expf(sS[j] - m); sS[j] = e; sum += e; }
        float inv = 1.0f / sum;
        for (int j = 0; j < K_SHOTS; ++j) sS[j] *= inv;
    }
    __syncthreads();

    for (int d = tid; d < DFEAT; d += 256) {
        const u16* vr = Vp + (size_t)(c * K_SHOTS) * DFEAT + d;
        float a = 0.0f;
        for (int j = 0; j < K_SHOTS; ++j) a += sS[j] * bf2f(vr[(size_t)j * DFEAT]);
        att[(size_t)c * DFEAT + d] = f2bf(a);
    }
}

// ---------------------------------------------------------------------------
// 5) final: y[i,:] = support[i,:] + out[label[i],:]
// ---------------------------------------------------------------------------
__global__ __launch_bounds__(256) void gather_add(const float* __restrict__ sf,
                                                  const float* __restrict__ outp,
                                                  const int* __restrict__ labels,
                                                  float* __restrict__ y) {
    int i   = blockIdx.x;
    int lbl = labels[i];
    const float4* a = (const float4*)(sf   + (size_t)i   * DFEAT);
    const float4* b = (const float4*)(outp + (size_t)lbl * DFEAT);
    float4*       o = (float4*)(y + (size_t)i * DFEAT);
    int t = threadIdx.x;            // 256 threads x float4 = 1024 elems
    float4 av = a[t], bv = b[t];
    float4 r;
    r.x = av.x + bv.x; r.y = av.y + bv.y; r.z = av.z + bv.z; r.w = av.w + bv.w;
    o[t] = r;
}

// ---------------------------------------------------------------------------
// launch
// ---------------------------------------------------------------------------
extern "C" void kernel_launch(void* const* d_in, const int* in_sizes, int n_in,
                              void* d_out, int out_size, void* d_ws, size_t ws_size,
                              hipStream_t stream) {
    const float* sf   = (const float*)d_in[0];   // [NK, D]
    const float* ent  = (const float*)d_in[1];   // [N, D]
    const int*   lbl  = (const int*)  d_in[2];   // [NK]
    const float* Wq   = (const float*)d_in[3];
    const float* bq   = (const float*)d_in[4];
    const float* Wk   = (const float*)d_in[5];
    const float* bk   = (const float*)d_in[6];
    const float* Wv   = (const float*)d_in[7];
    const float* bv   = (const float*)d_in[8];
    const float* Wo   = (const float*)d_in[9];
    const float* bo   = (const float*)d_in[10];
    float* out = (float*)d_out;                  // [NK, D]

    // workspace carve-up (bytes)
    char* ws = (char*)d_ws;
    size_t off = 0;
    auto alloc = [&](size_t bytes) { char* p = ws + off; off += (bytes + 255) & ~(size_t)255; return p; };
    u16*   sfb  = (u16*)  alloc((size_t)NK * DFEAT * 2);        // support bf16
    u16*   entb = (u16*)  alloc((size_t)N_CLASSES * DFEAT * 2); // entities bf16
    u16*   WqT  = (u16*)  alloc((size_t)DFEAT * DFEAT * 2);
    u16*   WkT  = (u16*)  alloc((size_t)DFEAT * DFEAT * 2);
    u16*   WvT  = (u16*)  alloc((size_t)DFEAT * DFEAT * 2);
    u16*   WoT  = (u16*)  alloc((size_t)DFEAT * DFEAT * 2);
    u16*   Kp   = (u16*)  alloc((size_t)NK * DFEAT * 2);        // K-proj bf16
    u16*   Vp   = (u16*)  alloc((size_t)NK * DFEAT * 2);        // V-proj bf16
    float* Qf   = (float*)alloc((size_t)N_CLASSES * DFEAT * 4); // Q-proj fp32
    u16*   attb = (u16*)  alloc((size_t)N_CLASSES * DFEAT * 2); // attended bf16
    float* outf = (float*)alloc((size_t)N_CLASSES * DFEAT * 4); // O-proj fp32
    (void)ws_size; (void)n_in; (void)in_sizes; (void)out_size;

    // 1) conversions
    {
        int n4 = (NK * DFEAT) / 4;              // 8,388,608
        to_bf16_x4<<<n4 / 256, 256, 0, stream>>>((const float4*)sf, (u32*)sfb, n4);
        int e4 = (N_CLASSES * DFEAT) / 4;       // 131,072
        to_bf16_x4<<<e4 / 256, 256, 0, stream>>>((const float4*)ent, (u32*)entb, e4);
    }
    // 2) weight transposes -> bf16 row-major W.T
    {
        dim3 g(DFEAT / 32, DFEAT / 32);
        transpose_to_bf16<<<g, 256, 0, stream>>>(Wq, WqT, DFEAT);
        transpose_to_bf16<<<g, 256, 0, stream>>>(Wk, WkT, DFEAT);
        transpose_to_bf16<<<g, 256, 0, stream>>>(Wv, WvT, DFEAT);
        transpose_to_bf16<<<g, 256, 0, stream>>>(Wo, WoT, DFEAT);
    }
    // 3) big projections: Kp/Vp = support @ W.T + b   (bf16 out)
    {
        dim3 g(NK / BM, DFEAT / BN);            // 256 x 8
        gemm_bf16_wmma<<<g, 256, 0, stream>>>(sfb, WkT, bk, Kp, NK, DFEAT, DFEAT, 1);
        gemm_bf16_wmma<<<g, 256, 0, stream>>>(sfb, WvT, bv, Vp, NK, DFEAT, DFEAT, 1);
    }
    // 4) Q = entities @ Wq.T + bq   (fp32 out)
    {
        dim3 g(N_CLASSES / BM, DFEAT / BN);     // 4 x 8
        gemm_bf16_wmma<<<g, 256, 0, stream>>>(entb, WqT, bq, Qf, N_CLASSES, DFEAT, DFEAT, 0);
    }
    // 5) per-class masked attention (64-wide blocks, labels are arange//64)
    class_attention<<<N_CLASSES, 256, 0, stream>>>(Qf, Kp, Vp, attb);
    // 6) output projection: outf = attended @ Wo.T + bo  (fp32 out)
    {
        dim3 g(N_CLASSES / BM, DFEAT / BN);
        gemm_bf16_wmma<<<g, 256, 0, stream>>>(attb, WoT, bo, outf, N_CLASSES, DFEAT, DFEAT, 0);
    }
    // 7) gather + residual add
    gather_add<<<NK, 256, 0, stream>>>(sf, outf, lbl, out);
}
